// Xcodec2ISTFT_10814727651759
// MI455X (gfx1250) — matrix operand order repaired and checked by reading, not compiled, not run on hardware
//
#include <hip/hip_runtime.h>

typedef __attribute__((ext_vector_type(16))) _Float16 v16h;
typedef __attribute__((ext_vector_type(8)))  float    v8f;

#define B_SZ    16
#define FREQ    513
#define FRAMES  2048
#define NFFT    1024
#define HOP     256
#define WIN     1024
#define KP      1024              // interleaved K: k'=2k (re), 2k+1 (im), k=0..511
#define KCHUNK  128               // K staged per LDS round
#define CHUNKS  (KP / KCHUNK)     // 8
#define M_TOTAL (B_SZ * FRAMES)   // 32768
#define OUTFULL ((FRAMES - 1) * HOP + WIN)   // 525056
#define PAD     ((WIN - HOP) / 2)            // 384
#define OUTLEN  (OUTFULL - 2 * PAD)          // 524288

// ---------------------------------------------------------------------------
// Kernel 1: windowed irfft basis, f16, K interleaved (re/im), transposed [n][k']
// Nyquist bin (k=512) handled in the GEMM epilogue as a rank-1 update.
// ---------------------------------------------------------------------------
__global__ void build_basis(const float* __restrict__ window,
                            _Float16* __restrict__ basis) {
    int idx = blockIdx.x * blockDim.x + threadIdx.x;   // over NFFT * KP
    if (idx >= NFFT * KP) return;
    int n  = idx / KP;
    int kp = idx % KP;
    int k  = kp >> 1;

    const float invN  = 1.0f / (float)NFFT;
    const float twoPi = 6.283185307179586f;
    int phase = (k * n) & (NFFT - 1);                  // exact arg reduction
    float ang = twoPi * (float)phase * invN;

    float c;
    if ((kp & 1) == 0) {                               // real-part slot
        float amp = (k == 0) ? invN : 2.0f * invN;
        c = amp * __cosf(ang);
    } else {                                           // imag-part slot
        c = (k == 0) ? 0.0f : -2.0f * invN * __sinf(ang);  // Im(DC) dropped
    }
    c *= window[n];                                    // fold synthesis window
    basis[idx] = (_Float16)c;
}

// ---------------------------------------------------------------------------
// Kernel 2: frames[32768,1024] = spec[.,K] x basis[K,.] via v_wmma f16->f32.
// Block = 256 threads = 8 wave32s; tile M=128 x N=64; wave strip 16x64.
// A: branch-free global loads (compile-time re/im parity, immediate offsets).
// B: 16KB LDS tile per 128-K round (barriers amortized over 16 WMMAs).
// ---------------------------------------------------------------------------
__global__ __launch_bounds__(256)
void istft_gemm(const float* __restrict__ re, const float* __restrict__ im,
                const float* __restrict__ window,
                const _Float16* __restrict__ basis, float* __restrict__ frames) {
    __shared__ __align__(64) _Float16 Bs[64 * KCHUNK];   // 16 KB, [n][k] 256B rows

    const int tid  = threadIdx.x;
    const int lane = tid & 31;
    const int wave = tid >> 5;
    const int n0   = blockIdx.x * 64;
    const int m0   = blockIdx.y * 128 + wave * 16;

    const int g  = lane >> 4;        // lane group within wave32
    const int lm = lane & 15;

    // A-fragment row: M = lane % 16 (ISA 7.12.2)
    const int m = m0 + lm;
    const int b = m / FRAMES;
    const int f = m % FRAMES;
    // per-lane base pointers; fold lane-group K offset (g*8 de-interleaved rows)
    const float* rp0 = re + (size_t)b * FREQ * FRAMES + f + (size_t)(g * 8) * FRAMES;
    const float* ip0 = im + (size_t)b * FREQ * FRAMES + f + (size_t)(g * 8) * FRAMES;

    v8f acc0 = {}, acc1 = {}, acc2 = {}, acc3 = {};

    // cooperative B copy: thread -> 64 contiguous bytes, fully coalesced
    const int cn = tid >> 2;            // 0..63  (n_local)
    const int cb = (tid & 3) * 32;      // half offset within 256B row

    for (int chunk = 0; chunk < CHUNKS; ++chunk) {
        const int kc = chunk * KCHUNK;

        __syncthreads();
        {
            const ulonglong2* src =
                (const ulonglong2*)(basis + (size_t)(n0 + cn) * KP + kc + cb);
            ulonglong2* dst = (ulonglong2*)(&Bs[cn * KCHUNK + cb]);
#pragma unroll
            for (int c = 0; c < 4; ++c) dst[c] = src[c];
        }
        __syncthreads();

#pragma unroll
        for (int ks = 0; ks < 4; ++ks) {
            const int kloc  = ks * 32;
            const int kbase = (kc + kloc) >> 1;      // de-interleaved k
            const float* rp = rp0 + (size_t)kbase * FRAMES;
            const float* ip = ip0 + (size_t)kbase * FRAMES;

            // A fragment: slot half*8+j <-> K = k0 + g*16 + half*8 + j;
            // parity(j) = re/im (compile-time), kk = kbase + g*8 + half*4 + j/2
            v16h afrag;
#pragma unroll
            for (int half = 0; half < 2; ++half) {
#pragma unroll
                for (int jp = 0; jp < 4; ++jp) {
                    afrag[half * 8 + 2 * jp]     = (_Float16)rp[(size_t)(half * 4 + jp) * FRAMES];
                    afrag[half * 8 + 2 * jp + 1] = (_Float16)ip[(size_t)(half * 4 + jp) * FRAMES];
                }
            }

            // B fragments: lane = column, 16 contiguous K halfs per lane group
#define BFRAG(j) (*(const v16h*)(&Bs[((j) * 16 + lm) * KCHUNK + kloc + g * 16]))
            acc0 = __builtin_amdgcn_wmma_f32_16x16x32_f16(false, afrag, false, BFRAG(0),
                                                          (short)0, acc0, false, false);
            acc1 = __builtin_amdgcn_wmma_f32_16x16x32_f16(false, afrag, false, BFRAG(1),
                                                          (short)0, acc1, false, false);
            acc2 = __builtin_amdgcn_wmma_f32_16x16x32_f16(false, afrag, false, BFRAG(2),
                                                          (short)0, acc2, false, false);
            acc3 = __builtin_amdgcn_wmma_f32_16x16x32_f16(false, afrag, false, BFRAG(3),
                                                          (short)0, acc3, false, false);
#undef BFRAG
        }
    }

    // ---- epilogue: Nyquist rank-1 update + store ---------------------------
    // x[n] += ReX[512] * (1/N) * cos(pi*n) * w[n];  cos(pi*n) = +/-1 by parity
    const float invN = 1.0f / (float)NFFT;
    const int   fb   = (m0 % FRAMES) + 8 * g;            // frame of C row r=0
    const float* re512 = re + (size_t)b * FREQ * FRAMES + (size_t)512 * FRAMES;
    const float sgn = (lm & 1) ? -invN : invN;           // n parity = lm parity
    float cw0 = window[n0 + lm]      * sgn;
    float cw1 = window[n0 + 16 + lm] * sgn;
    float cw2 = window[n0 + 32 + lm] * sgn;
    float cw3 = window[n0 + 48 + lm] * sgn;

    // C/D layout: VGPR r, lanes 0-15 -> M=r, lanes 16-31 -> M=r+8; N = lane%16
#pragma unroll
    for (int r = 0; r < 8; ++r) {
        const float a512 = re512[fb + r];
        float* dst = frames + (size_t)(m0 + 8 * g + r) * NFFT + n0 + lm;
        dst[0]  = acc0[r] + a512 * cw0;
        dst[16] = acc1[r] + a512 * cw1;
        dst[32] = acc2[r] + a512 * cw2;
        dst[48] = acc3[r] + a512 * cw3;
    }
}

// ---------------------------------------------------------------------------
// Kernel 3: overlap-add gather + window^2 envelope normalization + crop.
// Deterministic (no atomics): each output sample sums its <=4 source frames.
// ---------------------------------------------------------------------------
__global__ void ola_gather(const float* __restrict__ frames,
                           const float* __restrict__ window,
                           float* __restrict__ out) {
    size_t idx = (size_t)blockIdx.x * blockDim.x + threadIdx.x;
    if (idx >= (size_t)B_SZ * OUTLEN) return;
    const int b  = (int)(idx / OUTLEN);
    const int tp = (int)(idx % OUTLEN);
    const int t  = tp + PAD;

    int f_lo = (t >= (WIN - 1)) ? (t - (WIN - 1) + HOP - 1) / HOP : 0;
    int f_hi = t / HOP; if (f_hi > FRAMES - 1) f_hi = FRAMES - 1;

    float acc = 0.0f, env = 0.0f;
    for (int fr = f_lo; fr <= f_hi; ++fr) {
        const int n = t - fr * HOP;
        const float w = window[n];
        acc += frames[(size_t)(b * FRAMES + fr) * NFFT + n];
        env += w * w;
    }
    out[idx] = acc / env;
}

// ---------------------------------------------------------------------------
extern "C" void kernel_launch(void* const* d_in, const int* in_sizes, int n_in,
                              void* d_out, int out_size, void* d_ws, size_t ws_size,
                              hipStream_t stream) {
    const float* spec_real = (const float*)d_in[0];
    const float* spec_imag = (const float*)d_in[1];
    const float* window    = (const float*)d_in[2];
    // d_in[3..5] = n_fft / hop / win scalars (compile-time constants here)

    _Float16* basis  = (_Float16*)d_ws;                            // 2 MiB
    float*    frames = (float*)((char*)d_ws + (size_t)(4u << 20)); // 128 MiB

    {
        int total = NFFT * KP;
        build_basis<<<(total + 255) / 256, 256, 0, stream>>>(window, basis);
    }
    {
        dim3 grid(NFFT / 64, M_TOTAL / 128);   // N tiles fast -> L2 reuse of A
        istft_gemm<<<grid, 256, 0, stream>>>(spec_real, spec_imag, window,
                                             basis, frames);
    }
    {
        size_t total = (size_t)B_SZ * OUTLEN;
        ola_gather<<<(unsigned)((total + 255) / 256), 256, 0, stream>>>(
            frames, window, (float*)d_out);
    }
}
